// Decoder_78889959293085
// MI455X (gfx1250) — matrix operand order repaired and checked by reading, not compiled, not run on hardware
//
#include <hip/hip_runtime.h>
#include <stdint.h>

// ---------------------------------------------------------------------------
// Greedy 2-layer LSTM decoder, persistent-kernel implementation for gfx1250.
//
// Sequential chain of L-1 = 2047 steps. Per step (memory-bound, L2-resident):
//   Phase A: r0 = w_hh0@h0_prev, r1a = w_hh1@h1_prev          (WMMA bf16)
//   Phase B: cell0 (redundant per block) -> h0 ; r1b = w_ih1@h0 (WMMA bf16)
//   Phase C: cell1 -> h1 ; preds = relu(w_fc@h1 + b_fc) ; argmax -> token
// 3 grid barriers / step. Weights converted fp32->bf16 into d_ws once per
// call (~91 MB, fits the 192 MB L2 with room for the output stream).
// ---------------------------------------------------------------------------

#define H_  1024
#define G_  4096          // 4*H
#define V_  32000
#define L_  2048
#define NB_ 128           // persistent blocks (all must be co-resident)
#define NT_ 256           // threads per block = 8 wave32
#define NWAVES_ (NB_ * (NT_ / 32))   // 1024 waves total
#define TFC_ (V_ / 16)    // 2000 fc row-tiles

typedef __attribute__((ext_vector_type(16))) __bf16 v16bf;
typedef __attribute__((ext_vector_type(8)))  float  v8f;

union Frag { v16bf v; uint4 q[2]; };

struct DecParams {
  const int*   y;
  const float* ctx;
  const float* wih0;
  const float* bih0; const float* bhh0;
  const float* bih1; const float* bhh1;
  const float* bfc;
  const unsigned short* whh0b;
  const unsigned short* wih1b;
  const unsigned short* whh1b;
  const unsigned short* wfcb;
  float* gates0;
  float* gates1;
  float* token_f;
  unsigned* cnt;
  unsigned* gen;
  unsigned long long* slot;
  float* out;
};

__device__ __forceinline__ unsigned short f2bf(float f) {
  unsigned u = __float_as_uint(f);
  u += 0x7FFFu + ((u >> 16) & 1u);           // round-to-nearest-even
  return (unsigned short)(u >> 16);
}

__device__ __forceinline__ float sigmf(float x) {
  return 1.0f / (1.0f + __expf(-x));
}

// y16 = W[m:m+16, :] @ x  via V_WMMA_F32_16X16X32_BF16, B = x broadcast to
// all 16 columns (every column of D equals the matvec result).
// A layout (16-bit A, 16x32): lane half h=lane>>4 holds K runs {8h..8h+7} and
// {16+8h..16+8h+7}. B layout: same K runs as A.
__device__ __forceinline__ v8f mv16(const unsigned short* __restrict__ W,
                                    const unsigned short* __restrict__ x,
                                    int m) {
  const int lane = (int)(threadIdx.x & 31u);
  const int half = lane >> 4;
  const int row  = m + (lane & 15);
  const uint4* __restrict__ wp =
      reinterpret_cast<const uint4*>(W + (size_t)row * H_) + half;
  const uint4* __restrict__ xp =
      reinterpret_cast<const uint4*>(x) + half;
  v8f acc = {0.f, 0.f, 0.f, 0.f, 0.f, 0.f, 0.f, 0.f};
#pragma unroll 8
  for (int k = 0; k < H_ / 32; ++k) {
    Frag a, b;
    a.q[0] = wp[4 * k];          // K = 32k + 8*half + 0..7
    a.q[1] = wp[4 * k + 2];      // K = 32k + 8*half + 16..23
    b.q[0] = xp[4 * k];          // K = 32k + 8*half + 0..7
    b.q[1] = xp[4 * k + 2];      // K = 32k + 8*half + 16..23
    acc = __builtin_amdgcn_wmma_f32_16x16x32_bf16(
        false, a.v, false, b.v, (short)0, acc, false, false);
  }
  return acc;
}

// C/D layout: VGPR r, lanes 0-15 -> M=r col=lane; lanes 16-31 -> M=r+8.
// Column 0 lives in lanes 0 and 16; all columns are identical here.
__device__ __forceinline__ void store_rows(float* dst, int m, v8f a, int lane) {
  if ((lane & 15) == 0) {
    const int r = m + ((lane >> 4) << 3);
    *reinterpret_cast<float4*>(dst + r)     = make_float4(a[0], a[1], a[2], a[3]);
    *reinterpret_cast<float4*>(dst + r + 4) = make_float4(a[4], a[5], a[6], a[7]);
  }
}

__device__ __forceinline__ void add_rows(float* dst, int m, v8f a, int lane) {
  if ((lane & 15) == 0) {
    const int r = m + ((lane >> 4) << 3);
    float4 lo = *reinterpret_cast<float4*>(dst + r);
    float4 hi = *reinterpret_cast<float4*>(dst + r + 4);
    lo.x += a[0]; lo.y += a[1]; lo.z += a[2]; lo.w += a[3];
    hi.x += a[4]; hi.y += a[5]; hi.z += a[6]; hi.w += a[7];
    *reinterpret_cast<float4*>(dst + r)     = lo;
    *reinterpret_cast<float4*>(dst + r + 4) = hi;
  }
}

__device__ __forceinline__ void grid_barrier(unsigned* cnt, unsigned* gen) {
  __syncthreads();
  if (threadIdx.x == 0) {
    __threadfence();
    unsigned g = __hip_atomic_load(gen, __ATOMIC_RELAXED, __HIP_MEMORY_SCOPE_AGENT);
    unsigned prev =
        __hip_atomic_fetch_add(cnt, 1u, __ATOMIC_ACQ_REL, __HIP_MEMORY_SCOPE_AGENT);
    if (prev == NB_ - 1) {
      __hip_atomic_store(cnt, 0u, __ATOMIC_RELAXED, __HIP_MEMORY_SCOPE_AGENT);
      __hip_atomic_fetch_add(gen, 1u, __ATOMIC_RELEASE, __HIP_MEMORY_SCOPE_AGENT);
    } else {
      while (__hip_atomic_load(gen, __ATOMIC_ACQUIRE, __HIP_MEMORY_SCOPE_AGENT) == g) {
        __builtin_amdgcn_s_sleep(2);
      }
    }
  }
  __syncthreads();
  __threadfence();
}

__global__ void __launch_bounds__(NT_)
init_kernel(const float* whh0, const float* wih1, const float* whh1,
            const float* wfc,
            unsigned short* whh0b, unsigned short* wih1b,
            unsigned short* whh1b, unsigned short* wfcb,
            float* out, unsigned* cnt, unsigned* gen,
            unsigned long long* slot) {
  const size_t tid    = (size_t)blockIdx.x * blockDim.x + threadIdx.x;
  const size_t stride = (size_t)gridDim.x * blockDim.x;
  for (size_t i = tid; i < (size_t)G_ * H_; i += stride) {
    whh0b[i] = f2bf(whh0[i]);
    wih1b[i] = f2bf(wih1[i]);
    whh1b[i] = f2bf(whh1[i]);
  }
  for (size_t i = tid; i < (size_t)V_ * H_; i += stride) wfcb[i] = f2bf(wfc[i]);
  for (size_t i = tid; i < (size_t)V_; i += stride) out[i] = 0.0f;  // row 0
  if (tid == 0) { *cnt = 0u; *gen = 0u; *slot = 0ull; }
}

__global__ void __launch_bounds__(NT_)
decoder_kernel(DecParams p) {
  __shared__ alignas(16) unsigned short sh_h0[H_];
  __shared__ alignas(16) unsigned short sh_h1[H_];
  __shared__ float sh_c0[H_];
  __shared__ float sh_c1[H_];
  __shared__ unsigned long long sred[NT_];

  const int tid  = (int)threadIdx.x;
  const int lane = tid & 31;
  const int half = lane >> 4;
  const int gw   = (int)blockIdx.x * (NT_ / 32) + (tid >> 5);

  // h_s = h_c = context_vector initially; every block keeps a private copy.
#pragma unroll
  for (int j = 0; j < H_ / NT_; ++j) {
    const int i = tid + NT_ * j;
    const float h0 = p.ctx[i];
    const float h1 = p.ctx[H_ + i];
    sh_h0[i] = f2bf(h0); sh_c0[i] = h0;
    sh_h1[i] = f2bf(h1); sh_c1[i] = h1;
  }
  __syncthreads();

  for (int t = 1; t < L_; ++t) {
    // -------- Phase A: recurrent matvecs (depend only on previous step) ----
    if (gw < 256) {
      v8f a = mv16(p.whh0b, sh_h0, gw << 4);
      store_rows(p.gates0, gw << 4, a, lane);
    } else if (gw < 512) {
      v8f a = mv16(p.whh1b, sh_h1, (gw - 256) << 4);
      store_rows(p.gates1, (gw - 256) << 4, a, lane);
    }
    if (blockIdx.x == 0 && tid == 0) {
      float tok;
      if (t == 1) {
        tok = (float)p.y[0];
      } else {
        unsigned long long k = *p.slot;   // written at step t-1 phase C
        tok = (float)(0xFFFFFFFFu - (unsigned)(k & 0xFFFFFFFFull));
      }
      *p.token_f = tok;
      *p.slot = 0ull;                     // reset before this step's argmax
    }
    grid_barrier(p.cnt, p.gen);

    // -------- Phase B: cell0 (redundant per block), then w_ih1 @ h0 --------
    {
      const float tok = *p.token_f;
#pragma unroll
      for (int j = 0; j < H_ / NT_; ++j) {
        const int i = tid + NT_ * j;
        const float gi = p.gates0[i]          + p.wih0[i]          * tok + p.bih0[i]          + p.bhh0[i];
        const float gf = p.gates0[H_ + i]     + p.wih0[H_ + i]     * tok + p.bih0[H_ + i]     + p.bhh0[H_ + i];
        const float gg = p.gates0[2 * H_ + i] + p.wih0[2 * H_ + i] * tok + p.bih0[2 * H_ + i] + p.bhh0[2 * H_ + i];
        const float go = p.gates0[3 * H_ + i] + p.wih0[3 * H_ + i] * tok + p.bih0[3 * H_ + i] + p.bhh0[3 * H_ + i];
        const float ig = sigmf(gi), fg = sigmf(gf);
        const float g  = tanhf(gg), og = sigmf(go);
        const float cn = fg * sh_c0[i] + ig * g;
        sh_c0[i] = cn;
        sh_h0[i] = f2bf(og * tanhf(cn));
      }
    }
    __syncthreads();
    if (gw < 256) {
      v8f a = mv16(p.wih1b, sh_h0, gw << 4);
      add_rows(p.gates1, gw << 4, a, lane);   // gates1 already holds r1a
    }
    grid_barrier(p.cnt, p.gen);

    // -------- Phase C: cell1 (redundant), fc matvec + relu + argmax --------
#pragma unroll
    for (int j = 0; j < H_ / NT_; ++j) {
      const int i = tid + NT_ * j;
      const float gi = p.gates1[i]          + p.bih1[i]          + p.bhh1[i];
      const float gf = p.gates1[H_ + i]     + p.bih1[H_ + i]     + p.bhh1[H_ + i];
      const float gg = p.gates1[2 * H_ + i] + p.bih1[2 * H_ + i] + p.bhh1[2 * H_ + i];
      const float go = p.gates1[3 * H_ + i] + p.bih1[3 * H_ + i] + p.bhh1[3 * H_ + i];
      const float ig = sigmf(gi), fg = sigmf(gf);
      const float g  = tanhf(gg), og = sigmf(go);
      const float cn = fg * sh_c1[i] + ig * g;
      sh_c1[i] = cn;
      sh_h1[i] = f2bf(og * tanhf(cn));
    }
    __syncthreads();

    unsigned long long best = 0ull;
    float* outrow = p.out + (size_t)t * V_;
    for (int tile = gw; tile < TFC_; tile += NWAVES_) {
      v8f a = mv16(p.wfcb, sh_h1, tile << 4);
      if ((lane & 15) == 0) {
        const int r = (tile << 4) + (half << 3);
        float vv[8];
#pragma unroll
        for (int j = 0; j < 8; ++j) {
          float v = fmaxf(a[j] + p.bfc[r + j], 0.0f);  // relu
          vv[j] = v;
          // v >= 0 -> float bits are order-preserving; ~idx breaks ties low.
          unsigned long long key =
              ((unsigned long long)__float_as_uint(v) << 32) |
              (unsigned long long)(0xFFFFFFFFu - (unsigned)(r + j));
          if (key > best) best = key;
        }
        *reinterpret_cast<float4*>(outrow + r)     = make_float4(vv[0], vv[1], vv[2], vv[3]);
        *reinterpret_cast<float4*>(outrow + r + 4) = make_float4(vv[4], vv[5], vv[6], vv[7]);
      }
    }
    sred[tid] = best;
    __syncthreads();
    for (int s = NT_ / 2; s > 0; s >>= 1) {
      if (tid < s) {
        unsigned long long o = sred[tid + s];
        if (o > sred[tid]) sred[tid] = o;
      }
      __syncthreads();
    }
    if (tid == 0) atomicMax(p.slot, sred[0]);
    grid_barrier(p.cnt, p.gen);
  }
}

extern "C" void kernel_launch(void* const* d_in, const int* in_sizes, int n_in,
                              void* d_out, int out_size, void* d_ws, size_t ws_size,
                              hipStream_t stream) {
  (void)in_sizes; (void)n_in; (void)out_size; (void)ws_size;
  const int*   y    = (const int*)  d_in[0];
  const float* ctx  = (const float*)d_in[1];
  const float* wih0 = (const float*)d_in[2];
  const float* whh0 = (const float*)d_in[3];
  const float* bih0 = (const float*)d_in[4];
  const float* bhh0 = (const float*)d_in[5];
  const float* wih1 = (const float*)d_in[6];
  const float* whh1 = (const float*)d_in[7];
  const float* bih1 = (const float*)d_in[8];
  const float* bhh1 = (const float*)d_in[9];
  const float* wfc  = (const float*)d_in[10];
  const float* bfc  = (const float*)d_in[11];

  // Workspace layout (~90.8 MB): bf16 weights + tiny control state.
  char* ws = (char*)d_ws;
  size_t off = 0;
  auto take = [&](size_t bytes) -> char* {
    char* pp = ws + off;
    off = (off + bytes + 255) & ~(size_t)255;
    return pp;
  };
  unsigned short* whh0b = (unsigned short*)take((size_t)G_ * H_ * 2);
  unsigned short* wih1b = (unsigned short*)take((size_t)G_ * H_ * 2);
  unsigned short* whh1b = (unsigned short*)take((size_t)G_ * H_ * 2);
  unsigned short* wfcb  = (unsigned short*)take((size_t)V_ * H_ * 2);
  float* gates0  = (float*)take((size_t)G_ * 4);
  float* gates1  = (float*)take((size_t)G_ * 4);
  float* token_f = (float*)take(4);
  unsigned* cnt  = (unsigned*)take(4);
  unsigned* gen  = (unsigned*)take(4);
  unsigned long long* slot = (unsigned long long*)take(8);

  init_kernel<<<2048, NT_, 0, stream>>>(whh0, wih1, whh1, wfc,
                                        whh0b, wih1b, whh1b, wfcb,
                                        (float*)d_out, cnt, gen, slot);

  DecParams p;
  p.y = y; p.ctx = ctx; p.wih0 = wih0;
  p.bih0 = bih0; p.bhh0 = bhh0; p.bih1 = bih1; p.bhh1 = bhh1; p.bfc = bfc;
  p.whh0b = whh0b; p.wih1b = wih1b; p.whh1b = whh1b; p.wfcb = wfcb;
  p.gates0 = gates0; p.gates1 = gates1; p.token_f = token_f;
  p.cnt = cnt; p.gen = gen; p.slot = slot;
  p.out = (float*)d_out;

  decoder_kernel<<<NB_, NT_, 0, stream>>>(p);
}
